// GraphSAGE_77575699300503
// MI455X (gfx1250) — compile-verified, matrix operand order checked by reading, not run on hardware
//
#include <hip/hip_runtime.h>
#include <hip/hip_bf16.h>

// ---------------------------------------------------------------------------
// GraphSAGE (3 layers, mean aggregation) for MI455X / gfx1250, wave32.
//   per layer: msg[dst] += h[src] (fp32 HW atomics, resolved in 192MB L2),
//              h_neigh = msg * inv_deg, then
//              out = h @ Wself + h_neigh @ Wneigh + b  via v_wmma_f32_16x16x32_bf16
// ---------------------------------------------------------------------------

#define NN 50000
#define NE 600000
#define FD 128   // in/hidden feature dim for every layer input

typedef __attribute__((ext_vector_type(16))) __bf16          v16bf;
typedef __attribute__((ext_vector_type(8)))  float           v8f;
typedef __attribute__((ext_vector_type(8)))  unsigned short  us8;

union V16 { us8 h[2]; v16bf v; };

__device__ __forceinline__ unsigned short f2bf(float f) {
  unsigned int u = __float_as_uint(f);
  u += 0x7FFFu + ((u >> 16) & 1u);          // round-to-nearest-even
  return (unsigned short)(u >> 16);
}

// ---------------- utility kernels ----------------
__global__ void k_zero_f(float* __restrict__ p, int n) {
  int t = blockIdx.x * blockDim.x + threadIdx.x;
  if (t < n) p[t] = 0.0f;
}
__global__ void k_zero_i(int* __restrict__ p, int n) {
  int t = blockIdx.x * blockDim.x + threadIdx.x;
  if (t < n) p[t] = 0;
}

// degree (once, int atomics)
__global__ void k_deg(const int* __restrict__ dst, int* __restrict__ deg) {
  int t = blockIdx.x * blockDim.x + threadIdx.x;
  if (t < NE) atomicAdd(&deg[dst[t]], 1);
}
__global__ void k_invdeg(const int* __restrict__ deg, float* __restrict__ inv) {
  int t = blockIdx.x * blockDim.x + threadIdx.x;
  if (t < NN) inv[t] = 1.0f / fmaxf((float)deg[t], 1.0f);
}

// pack fp32 weight [K x N] (row-major) into the per-lane B-matrix bf16 layout:
// element index = ((nt*4 + ks)*32 + lane)*16 + e, N padded with zeros to 16*ntiles
__global__ void k_pack_w(const float* __restrict__ w, unsigned short* __restrict__ packed,
                         int K, int N, int ntilesN) {
  int t = blockIdx.x * blockDim.x + threadIdx.x;
  int total = ntilesN * 4 * 32 * 16;
  if (t >= total) return;
  int e    =  t        & 15;
  int lane = (t >> 4)  & 31;
  int ks   = (t >> 9)  & 3;
  int nt   =  t >> 11;
  int nl   = lane & 15;
  int kh   = (lane >> 4) << 3;                 // 0 or 8
  int klocal = ((e >> 3) << 4) + kh + (e & 7); // matches 16-bit A/B VGPR layout
  int k = (ks << 5) + klocal;
  int n = (nt << 4) + nl;
  float v = (k < K && n < N) ? w[k * N + n] : 0.0f;
  packed[t] = f2bf(v);
}

// edge scatter: msg[dst] += h[src]   (1 wave handles 1 edge row of 128 floats)
__global__ void k_scatter(const float* __restrict__ h, const int* __restrict__ src,
                          const int* __restrict__ dst, float* __restrict__ msg) {
  int t = blockIdx.x * blockDim.x + threadIdx.x;
  if (t >= NE * 32) return;
  int e    = t >> 5;
  int lane = t & 31;
  int f    = lane << 2;
  const float4 vin = *(const float4*)(h + (size_t)src[e] * FD + f);
  float* mp = msg + (size_t)dst[e] * FD + f;
  unsafeAtomicAdd(mp + 0, vin.x);   // -> global_atomic_add_f32 (L2-resident)
  unsafeAtomicAdd(mp + 1, vin.y);
  unsafeAtomicAdd(mp + 2, vin.z);
  unsafeAtomicAdd(mp + 3, vin.w);
}

// fused mean + bf16 convert of both GEMM inputs
__global__ void k_prep(const float* __restrict__ h, const float* __restrict__ msg,
                       const float* __restrict__ inv,
                       unsigned short* __restrict__ hbf, unsigned short* __restrict__ hnbf) {
  int t = blockIdx.x * blockDim.x + threadIdx.x;
  if (t >= NN * 32) return;
  int node = t >> 5;
  int f    = (t & 31) << 2;
  size_t o = (size_t)node * FD + f;
  float s  = inv[node];
  float4 hv = *(const float4*)(h + o);
  float4 mv = *(const float4*)(msg + o);
  hbf[o + 0] = f2bf(hv.x);  hbf[o + 1] = f2bf(hv.y);
  hbf[o + 2] = f2bf(hv.z);  hbf[o + 3] = f2bf(hv.w);
  hnbf[o + 0] = f2bf(mv.x * s);  hnbf[o + 1] = f2bf(mv.y * s);
  hnbf[o + 2] = f2bf(mv.z * s);  hnbf[o + 3] = f2bf(mv.w * s);
}

// WMMA GEMM: out[50000 x ncols] = hbf @ Wself + hnbf @ Wneigh + b (opt. ReLU)
// one wave per 16x16 output tile; 4 K-steps x 2 matmuls into one v8f accumulator
__global__ void k_gemm(const unsigned short* __restrict__ hbf,
                       const unsigned short* __restrict__ hnbf,
                       const unsigned short* __restrict__ pws,
                       const unsigned short* __restrict__ pwn,
                       const float* __restrict__ bias,
                       float* __restrict__ out,
                       int ntilesN, int ncols, int relu, int totalTiles) {
  int wave = (int)(threadIdx.x >> 5);
  int lane = (int)(threadIdx.x & 31);
  int tile = blockIdx.x * (blockDim.x >> 5) + wave;
  if (tile >= totalTiles) return;               // wave-uniform guard
  int nt = tile % ntilesN;
  int mt = tile / ntilesN;
  int row   = (mt << 4) + (lane & 15);
  int khalf = (lane >> 4) << 3;                 // 0 or 8

  const v16bf* bS = (const v16bf*)pws + (size_t)(nt * 4) * 32 + lane;
  const v16bf* bN = (const v16bf*)pwn + (size_t)(nt * 4) * 32 + lane;
  const unsigned short* aS = hbf  + (size_t)row * FD;
  const unsigned short* aN = hnbf + (size_t)row * FD;

  v8f acc = {};
  #pragma unroll
  for (int ks = 0; ks < 4; ++ks) {
    int k0 = (ks << 5) + khalf;
    V16 a, an;
    a.h[0]  = *(const us8*)(aS + k0);
    a.h[1]  = *(const us8*)(aS + k0 + 16);
    an.h[0] = *(const us8*)(aN + k0);
    an.h[1] = *(const us8*)(aN + k0 + 16);
    v16bf b  = bS[ks * 32];
    v16bf bn = bN[ks * 32];
    acc = __builtin_amdgcn_wmma_f32_16x16x32_bf16(false, a.v,  false, b,
                                                  (short)0, acc, false, false);
    acc = __builtin_amdgcn_wmma_f32_16x16x32_bf16(false, an.v, false, bn,
                                                  (short)0, acc, false, false);
  }

  int n = (nt << 4) + (lane & 15);
  if (n < ncols) {
    float bv = bias[n];
    #pragma unroll
    for (int r = 0; r < 8; ++r) {
      int m = (mt << 4) + r + khalf;
      float v = acc[r] + bv;
      if (relu) v = fmaxf(v, 0.0f);
      out[(size_t)m * ncols + n] = v;           // coalesced across lanes 0..15
    }
  }
}

// ---------------------------------------------------------------------------
static inline char* ws_take(char* base, size_t& off, size_t bytes) {
  off = (off + 255) & ~(size_t)255;
  char* p = base + off;
  off += bytes;
  return p;
}

static void run_layer(const float* hin, const unsigned short* pws, const unsigned short* pwn,
                      const float* bias, float* hout, int ncols, int ntilesN, int relu,
                      float* msg, const int* src, const int* dst, const float* invdeg,
                      unsigned short* hbf, unsigned short* hnbf, hipStream_t stream) {
  const int T = 256;
  k_zero_f<<<(NN * FD + T - 1) / T, T, 0, stream>>>(msg, NN * FD);
  k_scatter<<<(NE * 32 + T - 1) / T, T, 0, stream>>>(hin, src, dst, msg);
  k_prep<<<(NN * 32 + T - 1) / T, T, 0, stream>>>(hin, msg, invdeg, hbf, hnbf);
  int totalTiles = (NN / 16) * ntilesN;         // 50000 % 16 == 0
  int wavesPerBlk = 4;                          // 128 threads = 4 waves
  int blocks = (totalTiles + wavesPerBlk - 1) / wavesPerBlk;
  k_gemm<<<blocks, 32 * wavesPerBlk, 0, stream>>>(hbf, hnbf, pws, pwn, bias, hout,
                                                  ntilesN, ncols, relu, totalTiles);
}

extern "C" void kernel_launch(void* const* d_in, const int* in_sizes, int n_in,
                              void* d_out, int out_size, void* d_ws, size_t ws_size,
                              hipStream_t stream) {
  const float* feat = (const float*)d_in[0];
  const int*   src  = (const int*)d_in[1];
  const int*   dst  = (const int*)d_in[2];
  const float* wS[3] = { (const float*)d_in[3], (const float*)d_in[6], (const float*)d_in[9] };
  const float* wN[3] = { (const float*)d_in[4], (const float*)d_in[7], (const float*)d_in[10] };
  const float* bb[3] = { (const float*)d_in[5], (const float*)d_in[8], (const float*)d_in[11] };
  float* out = (float*)d_out;

  char* base = (char*)d_ws;
  size_t off = 0;
  int*            deg    = (int*)           ws_take(base, off, (size_t)NN * 4);
  float*          invdeg = (float*)         ws_take(base, off, (size_t)NN * 4);
  float*          msg    = (float*)         ws_take(base, off, (size_t)NN * FD * 4);
  float*          hA     = (float*)         ws_take(base, off, (size_t)NN * FD * 4);
  float*          hB     = (float*)         ws_take(base, off, (size_t)NN * FD * 4);
  unsigned short* hbf    = (unsigned short*)ws_take(base, off, (size_t)NN * FD * 2);
  unsigned short* hnbf   = (unsigned short*)ws_take(base, off, (size_t)NN * FD * 2);
  const int ntN[3]  = { 8, 8, 3 };             // 128,128,40(->48) output cols
  const int ncol[3] = { 128, 128, 40 };
  unsigned short* pS[3];
  unsigned short* pNg[3];
  for (int i = 0; i < 3; ++i) {
    size_t sz = (size_t)ntN[i] * 4 * 32 * 16 * 2;
    pS[i]  = (unsigned short*)ws_take(base, off, sz);
    pNg[i] = (unsigned short*)ws_take(base, off, sz);
  }

  const int T = 256;
  // degree -> inv_deg (once; identical for all layers)
  k_zero_i<<<(NN + T - 1) / T, T, 0, stream>>>(deg, NN);
  k_deg<<<(NE + T - 1) / T, T, 0, stream>>>(dst, deg);
  k_invdeg<<<(NN + T - 1) / T, T, 0, stream>>>(deg, invdeg);

  // pack the six small weight matrices into WMMA B-operand layout (bf16)
  for (int i = 0; i < 3; ++i) {
    int tot = ntN[i] * 4 * 32 * 16;
    k_pack_w<<<(tot + T - 1) / T, T, 0, stream>>>(wS[i], pS[i],  FD, ncol[i], ntN[i]);
    k_pack_w<<<(tot + T - 1) / T, T, 0, stream>>>(wN[i], pNg[i], FD, ncol[i], ntN[i]);
  }

  run_layer(feat, pS[0], pNg[0], bb[0], hA,  ncol[0], ntN[0], 1, msg, src, dst, invdeg, hbf, hnbf, stream);
  run_layer(hA,   pS[1], pNg[1], bb[1], hB,  ncol[1], ntN[1], 1, msg, src, dst, invdeg, hbf, hnbf, stream);
  run_layer(hB,   pS[2], pNg[2], bb[2], out, ncol[2], ntN[2], 0, msg, src, dst, invdeg, hbf, hnbf, stream);
}
